// CLAHEPreprocessingLayer_52072183496783
// MI455X (gfx1250) — matrix operand order repaired and checked by reading, not compiled, not run on hardware
//
#include <hip/hip_runtime.h>
#include <hip/hip_bf16.h>

// CLAHE: B=256, H=W=224, C=3, GRID=8 -> tiles 28x28 (area 784), NBINS=256,
// clip limit = max(1, int(3.0*784/256)) = 9.
//
// Pass 1: gray (u8) + per-tile 256-bin histogram (LDS ds_add atomics, one WG/tile)
// Pass 2: per-tile LUT. 256-bin prefix sum done as 16x16 matrix scan on the
//         CDNA5 matrix core: cdf = M*U + Lstrict*T via two chained
//         v_wmma_f32_16x16x32_f16 (exact: integer-valued f16 operands <= 784,
//         f32 accumulate). One wave32 per tile.
// Pass 3: per-pixel bilinear blend of 4 tile LUTs, write 3 channels.

#define NB 256
#define GRID_N 8
#define TH 28
#define TILE_AREA 784          // 28*28
#define N_TILES 16384          // 256 * 64
#define N_PIX 12845056         // 256*224*224
#define GRAY_BYTES 12845056    // u8 gray plane
#define CLIP_LIM 9u

typedef __attribute__((ext_vector_type(16))) _Float16 v16h;
typedef __attribute__((ext_vector_type(8)))  float    v8f;

// ---------------- Pass 1: gray + per-tile histogram ----------------
__global__ void clahe_hist_kernel(const float* __restrict__ x,
                                  unsigned char* __restrict__ gray,
                                  unsigned* __restrict__ hist) {
    __shared__ unsigned lh[NB];
    const int tid = threadIdx.x;
    lh[tid] = 0u;
    __syncthreads();

    const int t  = blockIdx.x;          // tile id = ((b*8)+ty)*8 + tx
    const int b  = t >> 6;
    const int ty = (t >> 3) & 7;
    const int tx = t & 7;

    for (int i = tid; i < TILE_AREA; i += 256) {
        const int py = i / TH;
        const int px = i - py * TH;
        const int y  = ty * TH + py;
        const int xx = tx * TH + px;
        const int idx = (b * 224 + y) * 224 + xx;
        const float* p = x + (long)idx * 3;
        float r  = fminf(fmaxf(floorf(p[0]), 0.0f), 255.0f);
        float g  = fminf(fmaxf(floorf(p[1]), 0.0f), 255.0f);
        float bl = fminf(fmaxf(floorf(p[2]), 0.0f), 255.0f);
        float gf = rintf(r * 0.299f + g * 0.587f + bl * 0.114f);  // RNE = jnp.round
        int gi = (int)fminf(fmaxf(gf, 0.0f), 255.0f);
        gray[idx] = (unsigned char)gi;
        atomicAdd(&lh[gi], 1u);
    }
    __syncthreads();
    hist[(long)t * NB + tid] = lh[tid];
}

// ---------------- Pass 2: LUT via WMMA matrix prefix-scan ----------------
// One wave32 per tile; 8 waves per block. histlut aliased: read u32 hist,
// write f32 LUT in place (each tile owned by exactly one wave).
//
// Layout assumptions (CDNA5 ISA 7.12.2):
//  A (16x32 f16): lane l -> row M = l&15; v16h elems 0..7 = K=(l<16?0:8)+j,
//                 elems 8..15 = K=16+(l<16?0:8)+j.
//  B (32x16 f16): lane l -> col N = l&15; same K order as A.
//  C/D (16x16 f32): VGPR i: lanes 0-15 -> M=i,N=lane; lanes 16-31 -> M=8+i.
__global__ void clahe_lut_kernel(unsigned* __restrict__ histlut) {
    const int lane = threadIdx.x & 31;
    const int tile = blockIdx.x * 8 + (threadIdx.x >> 5);
    const int r    = lane & 15;
    const bool hi  = lane >= 16;
    const int kb   = hi ? 8 : 0;

    // Load this lane's 8 contiguous hist bins: bins 16*r + kb .. +7  (32B aligned)
    const unsigned* hp = histlut + (long)tile * NB + r * 16 + kb;
    uint4 h0 = *(const uint4*)(hp);
    uint4 h1 = *(const uint4*)(hp + 4);
    unsigned hv[8] = {h0.x, h0.y, h0.z, h0.w, h1.x, h1.y, h1.z, h1.w};

    // A = [ min(hist,9) reshaped 16x16  |  strict-lower-triangular ones ]
    v16h a;
#pragma unroll
    for (int j = 0; j < 8; ++j) {
        unsigned c = hv[j]; if (c > CLIP_LIM) c = CLIP_LIM;
        a[j] = (_Float16)(float)c;
        const int jg = kb + j;                       // Lstrict column
        a[8 + j] = (_Float16)((jg < r) ? 1.0f : 0.0f);
    }

    // B1 = [ U (upper-tri ones, K<=N) ; 0 ]  -> row-wise inclusive prefix
    v16h b1;
#pragma unroll
    for (int e = 0; e < 16; ++e)
        b1[e] = (_Float16)((e < 8 && kb + e <= r) ? 1.0f : 0.0f);

    v8f acc = {};
    acc = __builtin_amdgcn_wmma_f32_16x16x32_f16(false, a, false, b1,
                                                 (short)0, acc, false, false);

    // Row totals t[row] = P[row][15]: col 15 lives in lane 15 (rows 0-7) and
    // lane 31 (rows 8-15). Broadcast to all lanes.
    float t[16];
#pragma unroll
    for (int i = 0; i < 8; ++i) {
        t[i]     = __shfl(acc[i], 15, 32);
        t[8 + i] = __shfl(acc[i], 31, 32);
    }

    // B2 = [ 0 ; T ] with T[k][n] = t[k]  -> adds exclusive scan of row totals
    v16h b2;
#pragma unroll
    for (int e = 0; e < 16; ++e)
        b2[e] = (e >= 8) ? (_Float16)t[kb + e - 8] : (_Float16)0.0f;

    acc = __builtin_amdgcn_wmma_f32_16x16x32_f16(false, a, false, b2,
                                                 (short)0, acc, false, false);

    // cdf[255] = total clipped mass; excess redistributed as excess*(bin+1)/256
    const float S = __shfl(acc[7], 31, 32);
    const float excess = (float)TILE_AREA - S;

    float* lut = (float*)histlut;   // overwrite hist in place
#pragma unroll
    for (int i = 0; i < 8; ++i) {
        const int row = hi ? (8 + i) : i;
        const int bin = row * 16 + r;                // col = lane&15
        float cdf = acc[i] + excess * (float)(bin + 1) * (1.0f / 256.0f);
        float v = rintf(cdf * (255.0f / 784.0f));
        v = fminf(fmaxf(v, 0.0f), 255.0f);
        lut[(long)tile * NB + bin] = v;
    }
}

// ---------------- Pass 3: bilinear LUT blend ----------------
__global__ void clahe_apply_kernel(const unsigned char* __restrict__ gray,
                                   const float* __restrict__ lut,
                                   float* __restrict__ out) {
    const int p = blockIdx.x * 256 + threadIdx.x;    // exact grid, no tail
    const int b   = p / 50176;                        // 224*224
    const int rem = p - b * 50176;
    const int y = rem / 224;
    const int x = rem - y * 224;
    const int g = (int)gray[p];

    const float fy = ((float)y + 0.5f) * (1.0f / 28.0f) - 0.5f;
    const float fx = ((float)x + 0.5f) * (1.0f / 28.0f) - 0.5f;
    const float y0f = floorf(fy), x0f = floorf(fx);
    const float wy = fy - y0f,    wx = fx - x0f;
    const int iy0 = min(max((int)y0f, 0), 7);
    const int iy1 = min(max((int)y0f + 1, 0), 7);
    const int ix0 = min(max((int)x0f, 0), 7);
    const int ix1 = min(max((int)x0f + 1, 0), 7);

    const float* lb = lut + (long)b * (64 * NB);
    const float v00 = lb[(iy0 * 8 + ix0) * NB + g];
    const float v01 = lb[(iy0 * 8 + ix1) * NB + g];
    const float v10 = lb[(iy1 * 8 + ix0) * NB + g];
    const float v11 = lb[(iy1 * 8 + ix1) * NB + g];

    const float top = v00 * (1.0f - wx) + v01 * wx;
    const float bot = v10 * (1.0f - wx) + v11 * wx;
    const float v   = top * (1.0f - wy) + bot * wy;

    float* o = out + (long)p * 3;
    o[0] = v; o[1] = v; o[2] = v;
}

extern "C" void kernel_launch(void* const* d_in, const int* in_sizes, int n_in,
                              void* d_out, int out_size, void* d_ws, size_t ws_size,
                              hipStream_t stream) {
    (void)in_sizes; (void)n_in; (void)out_size; (void)ws_size;
    const float* x = (const float*)d_in[0];
    float* out = (float*)d_out;

    unsigned char* gray = (unsigned char*)d_ws;
    unsigned* histlut = (unsigned*)((char*)d_ws + GRAY_BYTES);  // 16 MB, reused as f32 LUT

    clahe_hist_kernel <<<N_TILES,      256, 0, stream>>>(x, gray, histlut);
    clahe_lut_kernel  <<<N_TILES / 8,  256, 0, stream>>>(histlut);
    clahe_apply_kernel<<<N_PIX / 256,  256, 0, stream>>>(gray, (const float*)histlut, out);
}
